// Seq2SeqModel_64158221468131
// MI455X (gfx1250) — matrix operand order repaired and checked by reading, not compiled, not run on hardware
//
#include <hip/hip_runtime.h>
#include <hip/hip_bf16.h>
#include <stdint.h>
#include <stddef.h>

// Problem dims (from reference)
#define V_   32000
#define E_   256
#define H_   512
#define B_   32
#define SSRC 128
#define STGT 64
#define G_   2048   // 4*H

typedef __attribute__((ext_vector_type(16))) __bf16        v16bf;
typedef __attribute__((ext_vector_type(8)))  float          v8f;
typedef __attribute__((ext_vector_type(8)))  unsigned short ush8;
typedef __attribute__((ext_vector_type(16))) unsigned short ush16;
typedef __attribute__((ext_vector_type(4)))  unsigned int   u32x4;
typedef __attribute__((ext_vector_type(8)))  int            i32x8;
typedef __attribute__((ext_vector_type(4)))  int            i32x4;

#if defined(__AMDGCN__) && __has_builtin(__builtin_amdgcn_tensor_load_to_lds)
#define HAVE_TDM 1
#else
#define HAVE_TDM 0
#endif

union FragU {
  ush16 u16;
  ush8  u8[2];
  v16bf bf;
};

__device__ __forceinline__ unsigned short f2b(float x) {
  unsigned int u = __float_as_uint(x);
  unsigned int r = u + 0x7FFFu + ((u >> 16) & 1u);   // round-to-nearest-even
  return (unsigned short)(r >> 16);
}
__device__ __forceinline__ float b2f(unsigned short h) {
  return __uint_as_float(((unsigned int)h) << 16);
}
__device__ __forceinline__ float sigmoidf_(float x) {
  return 1.0f / (1.0f + expf(-x));
}

__device__ __forceinline__ void wait_tensorcnt0() {
#if defined(__AMDGCN__)
#if __has_builtin(__builtin_amdgcn_s_wait_tensorcnt)
  __builtin_amdgcn_s_wait_tensorcnt(0);
#else
  asm volatile("s_wait_tensorcnt 0x0" ::: "memory");
#endif
#endif
}

#if HAVE_TDM
// Issue one TDM 2D tile load: global (row-major, 2-byte elems) -> LDS.
// D# packing per CDNA5 ISA:
//   group0: [1:0]=count=1, [63:32]=lds_addr, [120:64]=global byte addr,
//           [127:126]=type=2
//   group1: [17:16]=data_size=1 (2B), [79:48]=tensor_dim0, [111:80]=tensor_dim1,
//           [127:112]=tile_dim0, [143:128]=tile_dim1, [207:160]=dim0_stride
//   groups 2/3 (+extra group on this toolchain): zero (<=2D tensor)
// This toolchain exposes the 6-arg builtin:
//   (u32x4 g0, i32x8 g1, i32x4, i32x4, i32x8, i32 cpol)
__device__ __forceinline__ void tdm_load_2d(unsigned lds_addr, const void* gp,
                                            unsigned tensor_d0, unsigned tensor_d1,
                                            unsigned stride0,
                                            unsigned tile_d0, unsigned tile_d1) {
  unsigned long long ga = (unsigned long long)(size_t)gp;
  u32x4 g0;
  g0[0] = 1u;                                              // count = 1
  g0[1] = lds_addr;                                        // lds byte offset
  g0[2] = (unsigned)ga;                                    // global addr lo
  g0[3] = (unsigned)((ga >> 32) & 0x1FFFFFFull) | (2u << 30); // hi + type=2

  i32x8 g1;
  g1[0] = (int)(1u << 16);                                 // data_size = 2B
  g1[1] = (int)((tensor_d0 & 0xFFFFu) << 16);              // dim0[15:0]
  g1[2] = (int)(((tensor_d0 >> 16) & 0xFFFFu) |
                ((tensor_d1 & 0xFFFFu) << 16));            // dim0[31:16]|dim1[15:0]
  g1[3] = (int)(((tensor_d1 >> 16) & 0xFFFFu) |
                ((tile_d0 & 0xFFFFu) << 16));              // dim1[31:16]|tile0
  g1[4] = (int)(tile_d1 & 0xFFFFu);                        // tile1; tile2=0
  g1[5] = (int)stride0;                                    // dim0_stride lo
  g1[6] = 0;                                               // stride hi / dim1_stride
  g1[7] = 0;

  i32x4 z4 = (i32x4)0;
  i32x8 z8 = (i32x8)0;
  __builtin_amdgcn_tensor_load_to_lds(g0, g1, z4, z4, z8, 0);
}
#endif

// ---------------------------------------------------------------------------
// Generic bf16 WMMA GEMM:  D(MxN) = A(MxK) * W(NxK)^T  [+ bias[n]] [+ Cin]
// One wave per 16x16 output tile (used for all small/medium GEMMs).
// ---------------------------------------------------------------------------
__global__ void k_gemm_bf16(const unsigned short* __restrict__ A, int lda,
                            const unsigned short* __restrict__ Bw, int ldb,
                            const float* __restrict__ bias,
                            const float* __restrict__ Cin, int ldc,
                            float* __restrict__ D, int ldd,
                            int M, int N, int K)
{
  const int lane = threadIdx.x & 31;
  const int wave = blockIdx.x * (blockDim.x >> 5) + (threadIdx.x >> 5);
  const int ntn  = N >> 4;
  const int total = (M >> 4) * ntn;
  if (wave >= total) return;                 // wave-uniform exit
  const int m0 = (wave / ntn) << 4;
  const int n0 = (wave % ntn) << 4;
  const int l  = lane & 15;
  const int hi = lane >> 4;

  v8f acc;
#pragma unroll
  for (int r = 0; r < 8; ++r) {
    float v = 0.0f;
    if (Cin)  v  = Cin[(size_t)(m0 + hi * 8 + r) * ldc + n0 + l];
    if (bias) v += bias[n0 + l];
    acc[r] = v;
  }

  const unsigned short* ap = A  + (size_t)(m0 + l) * lda;
  const unsigned short* bp = Bw + (size_t)(n0 + l) * ldb;

  for (int k0 = 0; k0 < K; k0 += 32) {
    FragU fa, fb;
    fa.u16   = *(const ush16*)(ap + k0 + hi * 16);
    fb.u16   = *(const ush16*)(bp + k0 + hi * 16);
    if (k0 + 32 < K) {
      __builtin_prefetch(ap + k0 + 32 + hi * 8, 0, 1);
      __builtin_prefetch(bp + k0 + 32 + hi * 16, 0, 1);
    }
    acc = __builtin_amdgcn_wmma_f32_16x16x32_bf16(
        false, fa.bf, false, fb.bf, (short)0, acc, false, false);
  }

#pragma unroll
  for (int r = 0; r < 8; ++r)
    D[(size_t)(m0 + hi * 8 + r) * ldd + n0 + l] = acc[r];
}

// ---------------------------------------------------------------------------
// Logits GEMM with TDM double-buffered LDS staging.
// Block = 256 threads (8 waves) computes a 32x256 macro-tile.
//   grid.x = N/256, grid.y = M/32.  Each wave: 2 m-tiles x 2 n-tiles.
// Per 64-wide K-chunk, wave 0 issues two tensor_load_to_lds (A 32x64,
// B 256x64), tracked with TENSORcnt; compute reads fragments from LDS.
// ---------------------------------------------------------------------------
#define TBM 32
#define TBN 256
#define TBK 64

__global__ void __launch_bounds__(256)
k_gemm_big(const unsigned short* __restrict__ A, int lda,   // (M,K)
           const unsigned short* __restrict__ Bw, int ldb,  // (N,K)
           const float* __restrict__ bias,
           float* __restrict__ D, int ldd,
           int M, int N, int K)
{
  __shared__ __align__(64) unsigned short sA[2][TBM * TBK];  // 2 * 4 KB
  __shared__ __align__(64) unsigned short sB[2][TBN * TBK];  // 2 * 32 KB

  const int lane = threadIdx.x & 31;
  const int wv   = threadIdx.x >> 5;
  const int l    = lane & 15;
  const int hi   = lane >> 4;

  const int m0 = blockIdx.y * TBM;
  const int n0 = blockIdx.x * TBN;
  const int nch = K / TBK;

  v8f acc00 = {0,0,0,0,0,0,0,0};  // [m-tile][n-tile]
  v8f acc01 = {0,0,0,0,0,0,0,0};
  v8f acc10 = {0,0,0,0,0,0,0,0};
  v8f acc11 = {0,0,0,0,0,0,0,0};

#if HAVE_TDM
  auto issue = [&](int ch, int buf) {
    tdm_load_2d((unsigned)(size_t)(void*)&sA[buf][0],
                A + (size_t)m0 * lda + ch * TBK,
                (unsigned)K, (unsigned)M, (unsigned)lda, TBK, TBM);
    tdm_load_2d((unsigned)(size_t)(void*)&sB[buf][0],
                Bw + (size_t)n0 * ldb + ch * TBK,
                (unsigned)K, (unsigned)N, (unsigned)ldb, TBK, TBN);
  };
#else
  auto stage = [&](int ch, int buf) {  // cooperative fallback copy
    for (int i = threadIdx.x; i < (TBM * TBK) / 8; i += 256) {
      int r = i >> 3, c = (i & 7) << 3;
      *(ush8*)&sA[buf][r * TBK + c] =
          *(const ush8*)(A + (size_t)(m0 + r) * lda + ch * TBK + c);
    }
    for (int i = threadIdx.x; i < (TBN * TBK) / 8; i += 256) {
      int r = i >> 3, c = (i & 7) << 3;
      *(ush8*)&sB[buf][r * TBK + c] =
          *(const ush8*)(Bw + (size_t)(n0 + r) * ldb + ch * TBK + c);
    }
  };
#endif

  auto compute = [&](int buf) {
    const unsigned short* a = &sA[buf][0];
    const unsigned short* b = &sB[buf][0];
#pragma unroll
    for (int kk = 0; kk < TBK; kk += 32) {
      FragU fa0, fa1, fb0, fb1;
      fa0.u16   = *(const ush16*)(a + (l)       * TBK + kk + hi * 16);
      fa1.u16   = *(const ush16*)(a + (16 + l)  * TBK + kk + hi * 16);
      fb0.u16   = *(const ush16*)(b + ((wv * 2 + 0) * 16 + l) * TBK + kk + hi * 16);
      fb1.u16   = *(const ush16*)(b + ((wv * 2 + 1) * 16 + l) * TBK + kk + hi * 16);
      acc00 = __builtin_amdgcn_wmma_f32_16x16x32_bf16(
          false, fa0.bf, false, fb0.bf, (short)0, acc00, false, false);
      acc10 = __builtin_amdgcn_wmma_f32_16x16x32_bf16(
          false, fa1.bf, false, fb0.bf, (short)0, acc10, false, false);
      acc01 = __builtin_amdgcn_wmma_f32_16x16x32_bf16(
          false, fa0.bf, false, fb1.bf, (short)0, acc01, false, false);
      acc11 = __builtin_amdgcn_wmma_f32_16x16x32_bf16(
          false, fa1.bf, false, fb1.bf, (short)0, acc11, false, false);
    }
  };

#if HAVE_TDM
  if (wv == 0) { issue(0, 0); wait_tensorcnt0(); }
  __syncthreads();
  for (int ch = 0; ch < nch; ++ch) {
    const int cur = ch & 1;
    if (wv == 0 && ch + 1 < nch) issue(ch + 1, cur ^ 1);
    compute(cur);
    if (wv == 0 && ch + 1 < nch) wait_tensorcnt0();
    __syncthreads();
  }
#else
  for (int ch = 0; ch < nch; ++ch) {
    stage(ch, 0);
    __syncthreads();
    compute(0);
    __syncthreads();
  }
#endif

  // Store D += bias  (C/D layout: VGPR r -> row m0+hi*8+r, col n0+l)
#pragma unroll
  for (int r = 0; r < 8; ++r) {
    int row0 = m0 + hi * 8 + r;
    int row1 = m0 + 16 + hi * 8 + r;
    int c0 = n0 + (wv * 2 + 0) * 16 + l;
    int c1 = n0 + (wv * 2 + 1) * 16 + l;
    D[(size_t)row0 * ldd + c0] = acc00[r] + bias[c0];
    D[(size_t)row0 * ldd + c1] = acc01[r] + bias[c1];
    D[(size_t)row1 * ldd + c0] = acc10[r] + bias[c0];
    D[(size_t)row1 * ldd + c1] = acc11[r] + bias[c1];
  }
}

// ---------------------------------------------------------------------------
// Elementwise / gather / pointwise kernels
// ---------------------------------------------------------------------------
__global__ void k_cvt_bf16(const float* __restrict__ src,
                           unsigned short* __restrict__ dst, int n) {
  for (int i = blockIdx.x * blockDim.x + threadIdx.x; i < n;
       i += gridDim.x * blockDim.x)
    dst[i] = f2b(src[i]);
}

__global__ void k_cvt_slice(const float* __restrict__ src, int sld, int soff,
                            unsigned short* __restrict__ dst, int dld, int doff,
                            int rows, int cols) {
  int n = rows * cols;
  for (int i = blockIdx.x * blockDim.x + threadIdx.x; i < n;
       i += gridDim.x * blockDim.x) {
    int r = i / cols, c = i - r * cols;
    dst[(size_t)r * dld + doff + c] = f2b(src[(size_t)r * sld + soff + c]);
  }
}

__global__ void k_embed(const int* __restrict__ tok,
                        const float* __restrict__ emb,
                        unsigned short* __restrict__ dst, int steps) {
  int n = steps * B_ * E_;
  for (int i = blockIdx.x * blockDim.x + threadIdx.x; i < n;
       i += gridDim.x * blockDim.x) {
    int e = i & (E_ - 1);
    int r = i >> 8;            // /E_
    int t = r / B_;
    int b = r - t * B_;
    int v = tok[b * steps + t];
    dst[i] = f2b(emb[(size_t)v * E_ + e]);
  }
}

__global__ void k_init_state(float* h, float* c, unsigned short* hb) {
  int i = blockIdx.x * blockDim.x + threadIdx.x;
  if (i < B_ * H_) { h[i] = 0.0f; c[i] = 0.0f; hb[i] = 0; }
}

__global__ void k_gates(const float* __restrict__ Z,
                        float* __restrict__ c, float* __restrict__ h,
                        unsigned short* __restrict__ hb,
                        unsigned short* __restrict__ seq_out,
                        int t, int seqlen) {
  int i = blockIdx.x * blockDim.x + threadIdx.x;
  if (i >= B_ * H_) return;
  int b = i / H_, j = i - b * H_;
  const float* z = Z + (size_t)b * G_;
  float zi = z[j], zf = z[H_ + j], zg = z[2 * H_ + j], zo = z[3 * H_ + j];
  float cn = sigmoidf_(zf) * c[i] + sigmoidf_(zi) * tanhf(zg);
  float hn = sigmoidf_(zo) * tanhf(cn);
  c[i] = cn;
  h[i] = hn;
  unsigned short hb16 = f2b(hn);
  hb[i] = hb16;
  seq_out[((size_t)b * seqlen + t) * H_ + j] = hb16;
}

__global__ void k_attn_ctx(const float* __restrict__ encProj,
                           const float* __restrict__ q,
                           const float* __restrict__ av,
                           const unsigned short* __restrict__ encOutB,
                           const unsigned short* __restrict__ hb,
                           unsigned short* __restrict__ x2) {
  __shared__ float sc[SSRC];
  __shared__ float red;
  int b = blockIdx.x;
  int s = threadIdx.x;

  const float* ep = encProj + ((size_t)b * SSRC + s) * H_;
  const float* qb = q + (size_t)b * H_;
  float e = 0.0f;
  for (int k = 0; k < H_; ++k)
    e += tanhf(ep[k] + qb[k]) * av[k];
  sc[s] = e;
  __syncthreads();
  if (s == 0) {
    float m = sc[0];
    for (int i = 1; i < SSRC; ++i) m = fmaxf(m, sc[i]);
    red = m;
  }
  __syncthreads();
  sc[s] = expf(sc[s] - red);
  __syncthreads();
  if (s == 0) {
    float sum = 0.0f;
    for (int i = 0; i < SSRC; ++i) sum += sc[i];
    red = sum;
  }
  __syncthreads();
  sc[s] = sc[s] / red;
  __syncthreads();

  unsigned short* xb = x2 + (size_t)b * (2 * H_);
  const unsigned short* eb0 = encOutB + (size_t)b * SSRC * H_;
  for (int j = s; j < H_; j += blockDim.x) {
    float acc = 0.0f;
    for (int ss = 0; ss < SSRC; ++ss)
      acc += sc[ss] * b2f(eb0[(size_t)ss * H_ + j]);
    xb[j]      = f2b(acc);                       // context half
    xb[H_ + j] = hb[(size_t)b * H_ + j];         // h half
  }
}

// ---------------------------------------------------------------------------
// Host side
// ---------------------------------------------------------------------------
static inline size_t alignup(size_t x) { return (x + 255) & ~(size_t)255; }

extern "C" void kernel_launch(void* const* d_in, const int* in_sizes, int n_in,
                              void* d_out, int out_size, void* d_ws, size_t ws_size,
                              hipStream_t stream) {
  (void)in_sizes; (void)n_in; (void)out_size; (void)ws_size;

  const int*   src     = (const int*)  d_in[0];
  const int*   tgt     = (const int*)  d_in[1];
  const float* enc_emb = (const float*)d_in[2];
  const float* enc_Wih = (const float*)d_in[3];
  const float* enc_Whh = (const float*)d_in[4];
  const float* enc_b   = (const float*)d_in[5];
  const float* attn_W  = (const float*)d_in[6];
  const float* attn_b  = (const float*)d_in[7];
  const float* attn_v  = (const float*)d_in[8];
  const float* dec_emb = (const float*)d_in[9];
  const float* dec_Wih = (const float*)d_in[10];
  const float* dec_Whh = (const float*)d_in[11];
  const float* dec_b   = (const float*)d_in[12];
  const float* out_W   = (const float*)d_in[13];
  const float* out_b   = (const float*)d_in[14];
  float*       out     = (float*)d_out;

  char* w = (char*)d_ws;
  size_t off = 0;
  auto take = [&](size_t bytes) -> void* {
    void* p = w + off;
    off = alignup(off + bytes);
    return p;
  };
  unsigned short* wEncWih  = (unsigned short*)take((size_t)G_ * E_ * 2);
  unsigned short* wEncWhh  = (unsigned short*)take((size_t)G_ * H_ * 2);
  unsigned short* wDecWihE = (unsigned short*)take((size_t)G_ * E_ * 2);
  unsigned short* wDecWc   = (unsigned short*)take((size_t)G_ * 2 * H_ * 2);
  unsigned short* wWe      = (unsigned short*)take((size_t)H_ * H_ * 2);
  unsigned short* wWh      = (unsigned short*)take((size_t)H_ * H_ * 2);
  unsigned short* wOut     = (unsigned short*)take((size_t)V_ * H_ * 2);
  unsigned short* Xe       = (unsigned short*)take((size_t)SSRC * B_ * E_ * 2);
  unsigned short* Xd       = (unsigned short*)take((size_t)STGT * B_ * E_ * 2);
  float*          ZXe      = (float*)take((size_t)SSRC * B_ * G_ * 4);
  float*          ZXd      = (float*)take((size_t)STGT * B_ * G_ * 4);
  float*          Zbuf     = (float*)take((size_t)B_ * G_ * 4);
  unsigned short* encOutB  = (unsigned short*)take((size_t)B_ * SSRC * H_ * 2);
  float*          encProj  = (float*)take((size_t)B_ * SSRC * H_ * 4);
  unsigned short* dechB    = (unsigned short*)take((size_t)B_ * STGT * H_ * 2);
  float*          hS       = (float*)take((size_t)B_ * H_ * 4);
  float*          cS       = (float*)take((size_t)B_ * H_ * 4);
  unsigned short* hb       = (unsigned short*)take((size_t)B_ * H_ * 2);
  float*          qS       = (float*)take((size_t)B_ * H_ * 4);
  unsigned short* x2       = (unsigned short*)take((size_t)B_ * 2 * H_ * 2);

  auto gemm = [&](const unsigned short* A, int lda,
                  const unsigned short* Bw, int ldb,
                  const float* bias, const float* Cin, int ldc,
                  float* D, int ldd, int M, int N, int K) {
    int tiles  = (M >> 4) * (N >> 4);
    int blocks = (tiles + 7) / 8;
    k_gemm_bf16<<<blocks, 256, 0, stream>>>(A, lda, Bw, ldb, bias, Cin, ldc,
                                            D, ldd, M, N, K);
  };

  // --- setup -----------------------------------------------------------------
  k_init_state<<<(B_ * H_ + 255) / 256, 256, 0, stream>>>(hS, cS, hb);

  k_cvt_bf16<<<1024, 256, 0, stream>>>(enc_Wih, wEncWih, G_ * E_);
  k_cvt_bf16<<<1024, 256, 0, stream>>>(enc_Whh, wEncWhh, G_ * H_);
  k_cvt_bf16<<<4096, 256, 0, stream>>>(out_W, wOut, V_ * H_);
  k_cvt_slice<<<1024, 256, 0, stream>>>(attn_W, 2 * H_, 0,   wWh, H_, 0, H_, H_);
  k_cvt_slice<<<1024, 256, 0, stream>>>(attn_W, 2 * H_, H_,  wWe, H_, 0, H_, H_);
  k_cvt_slice<<<1024, 256, 0, stream>>>(dec_Wih, E_ + H_, 0,  wDecWihE, E_, 0, G_, E_);
  k_cvt_slice<<<2048, 256, 0, stream>>>(dec_Wih, E_ + H_, E_, wDecWc, 2 * H_, 0,  G_, H_);
  k_cvt_slice<<<2048, 256, 0, stream>>>(dec_Whh, H_,      0,  wDecWc, 2 * H_, H_, G_, H_);

  k_embed<<<2048, 256, 0, stream>>>(src, enc_emb, Xe, SSRC);
  k_embed<<<1024, 256, 0, stream>>>(tgt, dec_emb, Xd, STGT);

  // --- input-projection GEMMs over all timesteps (WMMA) ----------------------
  gemm(Xe, E_, wEncWih,  E_, enc_b, nullptr, 0, ZXe, G_, SSRC * B_, G_, E_);
  gemm(Xd, E_, wDecWihE, E_, dec_b, nullptr, 0, ZXd, G_, STGT * B_, G_, E_);

  // --- encoder recurrence ----------------------------------------------------
  for (int t = 0; t < SSRC; ++t) {
    gemm(hb, H_, wEncWhh, H_, nullptr, ZXe + (size_t)t * B_ * G_, G_,
         Zbuf, G_, B_, G_, H_);
    k_gates<<<(B_ * H_ + 255) / 256, 256, 0, stream>>>(Zbuf, cS, hS, hb,
                                                       encOutB, t, SSRC);
  }

  // enc_proj = enc_out @ W_e^T
  gemm(encOutB, H_, wWe, H_, nullptr, nullptr, 0, encProj, H_,
       B_ * SSRC, H_, H_);

  // --- decoder recurrence with attention -------------------------------------
  for (int t = 0; t < STGT; ++t) {
    gemm(hb, H_, wWh, H_, attn_b, nullptr, 0, qS, H_, B_, H_, H_);
    k_attn_ctx<<<B_, SSRC, 0, stream>>>(encProj, qS, attn_v, encOutB, hb, x2);
    gemm(x2, 2 * H_, wDecWc, 2 * H_, nullptr, ZXd + (size_t)t * B_ * G_, G_,
         Zbuf, G_, B_, G_, 2 * H_);
    k_gates<<<(B_ * H_ + 255) / 256, 256, 0, stream>>>(Zbuf, cS, hS, hb,
                                                       dechB, t, STGT);
  }

  // --- logits: dominant GEMM, TDM-staged, double-buffered LDS ---------------
  {
    dim3 grid(V_ / TBN, (B_ * STGT) / TBM);   // (125, 64)
    k_gemm_big<<<grid, 256, 0, stream>>>(dechB, H_, wOut, H_, out_b,
                                         out, V_, B_ * STGT, V_, H_);
  }
}
